// FC_window_self_attention_15126874817244
// MI455X (gfx1250) — matrix operand order repaired and checked by reading, not compiled, not run on hardware
//
#include <hip/hip_runtime.h>

typedef unsigned short u16;
typedef __attribute__((ext_vector_type(16))) __bf16        v16bf;
typedef __attribute__((ext_vector_type(8)))  float         v8f;
typedef __attribute__((ext_vector_type(4)))  unsigned int  u32x4;
typedef __attribute__((ext_vector_type(4)))  float         f32x4;

// GCC-vector int4 pointer types matching the async-LDS builtin signature
typedef int v4i_gcc __attribute__((vector_size(16)));
typedef __attribute__((address_space(1))) v4i_gcc* gas1_i4p;
typedef __attribute__((address_space(3))) v4i_gcc* las3_i4p;

union FragU { u32x4 u[2]; v16bf v; };

// ---- constants ----
#define D_     768
#define NH     24
#define HD     32
#define NTOK   49
#define NW_    64
#define BW     2048
#define MROWS  (BW * NTOK)          // 100352
#define SCALE_ 0.17677669529663687f // 32^-0.5

#ifndef __has_builtin
#define __has_builtin(x) 0
#endif

#if __has_builtin(__builtin_amdgcn_global_load_async_to_lds_b128)
#define HAVE_ASYNC 1
#else
#define HAVE_ASYNC 0
#endif

#if HAVE_ASYNC
#if __has_builtin(__builtin_amdgcn_s_wait_asynccnt)
#define WAIT_ASYNC(n) __builtin_amdgcn_s_wait_asynccnt(n)
#else
#define WAIT_ASYNC(n) asm volatile("s_wait_asynccnt %0" ::"n"(n) : "memory")
#endif
#else
#define WAIT_ASYNC(n) ((void)0)
#endif

// one 16-byte chunk: global -> LDS (async when available)
__device__ __forceinline__ void ld_b128_to_lds(const u16* g, u16* l) {
#if HAVE_ASYNC
    __builtin_amdgcn_global_load_async_to_lds_b128(
        (gas1_i4p)(void*)(const_cast<u16*>(g)),
        (las3_i4p)(void*)l, 0, 0);
#else
    *(u32x4*)l = *(const u32x4*)g;
#endif
}

// stage 4 consecutive 16B chunks (32 bf16) of one row
__device__ __forceinline__ void stage_row64(const u16* g, u16* l) {
#pragma unroll
    for (int j = 0; j < 4; ++j) ld_b128_to_lds(g + j * 8, l + j * 8);
}

__device__ __forceinline__ u16 f2bf(float f) {
    unsigned int u = __float_as_uint(f);
    unsigned int r = u + 0x7FFFu + ((u >> 16) & 1u);
    return (u16)(r >> 16);
}

// ---------------- fp32 -> bf16 convert (8 elems / thread) ----------------
__global__ __launch_bounds__(256) void cvt_f32_bf16(const float* __restrict__ in,
                                                    u16* __restrict__ out, long n) {
    long i = ((long)blockIdx.x * 256 + threadIdx.x) * 8;
    if (i + 8 > n) return;
    f32x4 a = *(const f32x4*)(in + i);
    f32x4 b = *(const f32x4*)(in + i + 4);
    u16 tmp[8];
    tmp[0] = f2bf(a.x); tmp[1] = f2bf(a.y); tmp[2] = f2bf(a.z); tmp[3] = f2bf(a.w);
    tmp[4] = f2bf(b.x); tmp[5] = f2bf(b.y); tmp[6] = f2bf(b.z); tmp[7] = f2bf(b.w);
    *(u32x4*)(out + i) = *(u32x4*)tmp;
}

// ---------------- fp32 W[K][N] -> bf16 WT[N][K] ----------------
__global__ __launch_bounds__(256) void transpose_to_bf16(const float* __restrict__ W,
                                                         u16* __restrict__ WT,
                                                         int K, int N) {
    long idx = (long)blockIdx.x * 256 + threadIdx.x;
    if (idx >= (long)K * N) return;
    long k = idx / N, n = idx % N;
    WT[n * (long)K + k] = f2bf(W[idx]);
}

// ---------------- bf16 WMMA GEMM: C[M][N] = A[M][K] * BT[N][K]^T + bias ----------------
// 256 threads = 8 waves. Block tile 128x128. Wave tile 64(M) x 32(N).
// K-step 64, double-buffered LDS filled by async global->LDS loads.
template <bool OUT_BF16>
__global__ __launch_bounds__(256)
void gemm_bf16(const u16* __restrict__ A, const u16* __restrict__ BT,
               const float* __restrict__ bias, void* __restrict__ Cout,
               int M, int N, int K) {
    __shared__ u16 As[2][128][72]; // 144B row stride: 16B aligned, bank-spread
    __shared__ u16 Bs[2][128][72];

    const int tid  = threadIdx.x;
    const int lane = tid & 31;
    const int wave = tid >> 5;
    const int wm   = wave & 1;       // 2 M-slots of 64 rows
    const int wn   = wave >> 1;      // 4 N-slots of 32 cols
    const int l16  = lane & 15;
    const int hh   = lane >> 4;
    const long m0  = (long)blockIdx.y * 128;
    const long n0  = (long)blockIdx.x * 128;

    v8f acc[4][2];
#pragma unroll
    for (int mt = 0; mt < 4; ++mt)
#pragma unroll
        for (int nt = 0; nt < 2; ++nt)
#pragma unroll
            for (int r = 0; r < 8; ++r) acc[mt][nt][r] = 0.0f;

    // staging: 128 rows x 64 bf16; 2 threads per row, 64B each
    const int srow = tid >> 1;            // 0..127
    const int scol = (tid & 1) * 32;      // bf16 col offset

    const u16* gA = A  + (m0 + srow) * K + scol;
    const u16* gB = BT + (n0 + srow) * K + scol;

    // prologue: stage k-slab 0 into buffer 0 (8 async loads / thread)
    stage_row64(gA, &As[0][srow][scol]);
    stage_row64(gB, &Bs[0][srow][scol]);

    const int nk = K / 64;
    for (int ki = 0; ki < nk; ++ki) {
        const int cur = ki & 1;
        if (ki + 1 < nk) {
            const int nxt = cur ^ 1;
            stage_row64(gA + (ki + 1) * 64, &As[nxt][srow][scol]);
            stage_row64(gB + (ki + 1) * 64, &Bs[nxt][srow][scol]);
            WAIT_ASYNC(8);   // allow the 8 just-issued to remain in flight
        } else {
            WAIT_ASYNC(0);
        }
        __syncthreads();

#pragma unroll
        for (int kh = 0; kh < 2; ++kh) {
            FragU af[4];
#pragma unroll
            for (int mt = 0; mt < 4; ++mt) {
                int row = wm * 64 + mt * 16 + l16;
                af[mt].u[0] = *(const u32x4*)&As[cur][row][kh * 32 + 8 * hh];
                af[mt].u[1] = *(const u32x4*)&As[cur][row][kh * 32 + 16 + 8 * hh];
            }
            FragU bfr[2];
#pragma unroll
            for (int nt = 0; nt < 2; ++nt) {
                int col = wn * 32 + nt * 16 + l16;
                bfr[nt].u[0] = *(const u32x4*)&Bs[cur][col][kh * 32 + 8 * hh];
                bfr[nt].u[1] = *(const u32x4*)&Bs[cur][col][kh * 32 + 16 + 8 * hh];
            }
#pragma unroll
            for (int mt = 0; mt < 4; ++mt)
#pragma unroll
                for (int nt = 0; nt < 2; ++nt)
                    acc[mt][nt] = __builtin_amdgcn_wmma_f32_16x16x32_bf16(
                        false, af[mt].v, false, bfr[nt].v, (short)0, acc[mt][nt],
                        false, false);
        }
        __syncthreads();
    }

#pragma unroll
    for (int nt = 0; nt < 2; ++nt) {
        long n = n0 + wn * 32 + nt * 16 + l16;
        float bv = bias[n];
#pragma unroll
        for (int mt = 0; mt < 4; ++mt) {
#pragma unroll
            for (int r = 0; r < 8; ++r) {
                long m = m0 + wm * 64 + mt * 16 + hh * 8 + r;
                float v = acc[mt][nt][r] + bv;
                if (OUT_BF16) ((u16*)Cout)[m * N + n] = f2bf(v);
                else          ((float*)Cout)[m * N + n] = v;
            }
        }
    }
}

// ---------------- window attention: one block per (window b, head) ----------------
__global__ __launch_bounds__(128)
void attn_kernel(const u16* __restrict__ qkv, const float* __restrict__ mask,
                 const float* __restrict__ bias_table, u16* __restrict__ att) {
    __shared__ u16  q_lds[64][40];
    __shared__ u16  k_lds[64][40];
    __shared__ u16  vt_lds[32][72];
    __shared__ float s_lds[64][68];
    __shared__ u16  p_lds[64][72];

    const int b    = blockIdx.x;
    const int head = blockIdx.y;
    const int tid  = threadIdx.x;
    const int lane = tid & 31;
    const int wave = tid >> 5;
    const int l16  = lane & 15;
    const int hh   = lane >> 4;

    // ---- stage q,k (row-major, async) and v (transposed, zero-padded) ----
    for (int i = tid; i < 256; i += 128) {
        int token = i >> 2;
        int c     = (i & 3) * 8;
        const u16* row = qkv + ((long)(b * NTOK + token)) * (3 * D_) + head * HD + c;
        if (token < NTOK) {
            ld_b128_to_lds(row,      &q_lds[token][c]);
            ld_b128_to_lds(row + D_, &k_lds[token][c]);
        }
        // v must be exactly zero in padding rows (P*NaN would poison real outputs)
        u32x4 v4 = {};
        if (token < NTOK) v4 = *(const u32x4*)(row + 2 * D_);
        const u16* pv = (const u16*)&v4;
#pragma unroll
        for (int e = 0; e < 8; ++e) vt_lds[c + e][token] = pv[e];
    }
    WAIT_ASYNC(0);
    __syncthreads();

    // ---- S = (Q @ K^T) * scale ; each wave owns one 16-row tile ----
    {
        FragU a;
        int row = wave * 16 + l16;
        a.u[0] = *(const u32x4*)&q_lds[row][8 * hh];
        a.u[1] = *(const u32x4*)&q_lds[row][16 + 8 * hh];
#pragma unroll
        for (int nt = 0; nt < 4; ++nt) {
            FragU bfr;
            int col = nt * 16 + l16;
            bfr.u[0] = *(const u32x4*)&k_lds[col][8 * hh];
            bfr.u[1] = *(const u32x4*)&k_lds[col][16 + 8 * hh];
            v8f s;
#pragma unroll
            for (int r = 0; r < 8; ++r) s[r] = 0.0f;
            s = __builtin_amdgcn_wmma_f32_16x16x32_bf16(false, a.v, false, bfr.v,
                                                        (short)0, s, false, false);
#pragma unroll
            for (int r = 0; r < 8; ++r)
                s_lds[wave * 16 + hh * 8 + r][nt * 16 + l16] = s[r] * SCALE_;
        }
    }
    __syncthreads();

    // ---- softmax rows (relative-position bias + window mask) ----
    if (tid < 64) {
        int i = tid;
        if (i < NTOK) {
            int ih = i / 7, iw = i % 7;
            const float* mrow = mask + ((long)(b & (NW_ - 1)) * NTOK + i) * NTOK;
            float mx = -1e30f;
            for (int j = 0; j < NTOK; ++j) {
                int jh = j / 7, jw = j % 7;
                int ridx = (ih - jh + 6) * 13 + (iw - jw + 6);
                float t = s_lds[i][j] + bias_table[ridx * NH + head] + mrow[j];
                s_lds[i][j] = t;
                mx = fmaxf(mx, t);
            }
            float sum = 0.0f;
            for (int j = 0; j < NTOK; ++j) {
                float e = __expf(s_lds[i][j] - mx);
                s_lds[i][j] = e;
                sum += e;
            }
            float inv = 1.0f / sum;
            for (int j = 0; j < NTOK; ++j) p_lds[i][j] = f2bf(s_lds[i][j] * inv);
            for (int j = NTOK; j < 64; ++j) p_lds[i][j] = 0;
        } else {
            for (int j = 0; j < 64; ++j) p_lds[i][j] = 0;
        }
    }
    __syncthreads();

    // ---- O = P @ V ----
    v8f o[2];
#pragma unroll
    for (int nt = 0; nt < 2; ++nt)
#pragma unroll
        for (int r = 0; r < 8; ++r) o[nt][r] = 0.0f;

    {
        int row = wave * 16 + l16;
#pragma unroll
        for (int kk = 0; kk < 64; kk += 32) {
            FragU a;
            a.u[0] = *(const u32x4*)&p_lds[row][kk + 8 * hh];
            a.u[1] = *(const u32x4*)&p_lds[row][kk + 16 + 8 * hh];
#pragma unroll
            for (int nt = 0; nt < 2; ++nt) {
                FragU bfr;
                int nd = nt * 16 + l16;
                bfr.u[0] = *(const u32x4*)&vt_lds[nd][kk + 8 * hh];
                bfr.u[1] = *(const u32x4*)&vt_lds[nd][kk + 16 + 8 * hh];
                o[nt] = __builtin_amdgcn_wmma_f32_16x16x32_bf16(false, a.v, false, bfr.v,
                                                                (short)0, o[nt], false, false);
            }
        }
    }
#pragma unroll
    for (int nt = 0; nt < 2; ++nt) {
#pragma unroll
        for (int r = 0; r < 8; ++r) {
            int token = wave * 16 + hh * 8 + r;
            if (token < NTOK)
                att[((long)(b * NTOK + token)) * D_ + head * HD + nt * 16 + l16] =
                    f2bf(o[nt][r]);
        }
    }
}

// ---------------- workspace layout (bytes, 256-aligned) ----------------
#define OFF_XBF   0L
#define OFF_QKVWT 154140672L
#define OFF_PRJWT 157679616L
#define OFF_QKV   158859264L
#define OFF_ATT   621281280L

extern "C" void kernel_launch(void* const* d_in, const int* in_sizes, int n_in,
                              void* d_out, int out_size, void* d_ws, size_t ws_size,
                              hipStream_t stream) {
    const float* x          = (const float*)d_in[0];
    const float* mask       = (const float*)d_in[1];
    const float* qkv_w      = (const float*)d_in[2];
    const float* qkv_b      = (const float*)d_in[3];
    const float* proj_w     = (const float*)d_in[4];
    const float* proj_b     = (const float*)d_in[5];
    const float* bias_table = (const float*)d_in[6];

    char* ws = (char*)d_ws;
    u16* x_bf  = (u16*)(ws + OFF_XBF);
    u16* qkvT  = (u16*)(ws + OFF_QKVWT);
    u16* projT = (u16*)(ws + OFF_PRJWT);
    u16* qkv   = (u16*)(ws + OFF_QKV);
    u16* att   = (u16*)(ws + OFF_ATT);

    const long nx = (long)MROWS * D_;               // 77,070,336
    cvt_f32_bf16<<<(unsigned)(nx / 8 / 256), 256, 0, stream>>>(x, x_bf, nx);

    transpose_to_bf16<<<(768 * 2304 + 255) / 256, 256, 0, stream>>>(qkv_w, qkvT, D_, 3 * D_);
    transpose_to_bf16<<<(768 * 768 + 255) / 256, 256, 0, stream>>>(proj_w, projT, D_, D_);

    // QKV GEMM: 100352 x 2304, K=768
    gemm_bf16<true><<<dim3(3 * D_ / 128, MROWS / 128), 256, 0, stream>>>(
        x_bf, qkvT, qkv_b, qkv, MROWS, 3 * D_, D_);

    // attention per (window, head)
    attn_kernel<<<dim3(BW, NH), 128, 0, stream>>>(qkv, mask, bias_table, att);

    // proj GEMM: 100352 x 768, K=768 -> fp32 output
    gemm_bf16<false><<<dim3(D_ / 128, MROWS / 128), 256, 0, stream>>>(
        att, projT, proj_b, d_out, MROWS, D_, D_);
}
